// NTM_89704686944577
// MI455X (gfx1250) — compile-verified, matrix operand order checked by reading, not run on hardware
//
#include <hip/hip_runtime.h>
#include <hip/hip_bf16.h>
#include <stdint.h>

// ---------------- sizes (fixed by the reference) ----------------
#define BB     64     // batch
#define SS     128    // seq len
#define DD     128    // embed dim
#define VV     512    // vocab
#define CC     256    // controller dim
#define NN     256    // memory slots
#define MM     64     // memory width
#define RB     16     // batch rows per workgroup (== WMMA M tile)
#define NGRP   4      // BB / RB workgroups
#define THREADS 256   // 8 wave32

typedef __attribute__((ext_vector_type(16))) __bf16 v16bf;
typedef __attribute__((ext_vector_type(8)))  float  v8f;

// ---------------- helpers ----------------
__device__ __forceinline__ uint32_t f2bf1(float f) {
  uint32_t u = __float_as_uint(f);
  u += 0x7fffu + ((u >> 16) & 1u);           // round-to-nearest-even
  return u >> 16;
}
__device__ __forceinline__ uint32_t packbf(float lo, float hi) {
  return (f2bf1(hi) << 16) | f2bf1(lo);
}
// 16-bit A/B matrix K mapping per CDNA5 WMMA VGPR layout (16x32 tile):
// lanes 0-15 hold K 0..7 (vgpr0-3) and 16..23 (vgpr4-7); lanes 16-31 add +8.
__device__ __forceinline__ int kmap(int l, int j) {
  int base = (l & 16) ? 8 : 0;
  return base + ((j < 4) ? (2 * j) : (16 + 2 * (j - 4)));
}
__device__ __forceinline__ float sigf(float x)      { return 1.f / (1.f + __expf(-x)); }
__device__ __forceinline__ float softplusf(float x) { return logf(1.f + __expf(x)); }

__device__ __forceinline__ v8f wmma_step(v8f acc, const uint32_t* aBlk,
                                         const uint32_t* bBlk, int lane) {
  v16bf a = *(const v16bf*)(aBlk + lane * 8);   // 32B per lane, 32B aligned
  v16bf b = *(const v16bf*)(bBlk + lane * 8);
  return __builtin_amdgcn_wmma_f32_16x16x32_bf16(false, a, false, b,
                                                 (short)0, acc, false, false);
}

// ---------------- prep kernels ----------------
__global__ void copy_mem_kernel(const float* __restrict__ src,
                                float* __restrict__ dst, int n) {
  for (int i = blockIdx.x * blockDim.x + threadIdx.x; i < n;
       i += gridDim.x * blockDim.x)
    dst[i] = src[i];
}

// Pack W (row-major [Nout][K]) as W^T into WMMA-B tile blocks:
// block (kt,nt) = 32 lanes x 8 dwords (16 bf16/lane), blockIdx.x = kt*NT+nt.
__global__ void pack_w_kernel(const float* __restrict__ W, uint32_t* __restrict__ dst,
                              int K, int Nout, int NT) {
  int blk = blockIdx.x;
  int kt = blk / NT, nt = blk % NT;
  int t = threadIdx.x, l = t >> 3, j = t & 7;
  int k = kt * 32 + kmap(l, j);
  int n = nt * 16 + (l & 15);
  float v0 = (n < Nout && k     < K) ? W[(size_t)n * K + k]     : 0.f;
  float v1 = (n < Nout && k + 1 < K) ? W[(size_t)n * K + k + 1] : 0.f;
  dst[(size_t)blk * 256 + l * 8 + j] = packbf(v0, v1);
}

// Concatenated head matrix Whead[198][256]:
// rows 0-63 key, 64 beta, 65 gate, 66-68 shift, 69 gamma, 70-133 erase, 134-197 add.
__device__ __forceinline__ float head_fetch(int n, int k,
    const float* keyW, const float* betaW, const float* gateW,
    const float* shiftW, const float* gammaW, const float* eraseW, const float* addW) {
  if (n < 64)   return keyW[n * CC + k];
  if (n == 64)  return betaW[k];
  if (n == 65)  return gateW[k];
  if (n < 69)   return shiftW[(n - 66) * CC + k];
  if (n == 69)  return gammaW[k];
  if (n < 134)  return eraseW[(n - 70) * CC + k];
  if (n < 198)  return addW[(n - 134) * CC + k];
  return 0.f;
}
__global__ void pack_head_kernel(const float* keyW, const float* betaW,
                                 const float* gateW, const float* shiftW,
                                 const float* gammaW, const float* eraseW,
                                 const float* addW, uint32_t* __restrict__ dst) {
  const int NT = 13;                       // 208 padded cols
  int blk = blockIdx.x;
  int kt = blk / NT, nt = blk % NT;
  int t = threadIdx.x, l = t >> 3, j = t & 7;
  int k = kt * 32 + kmap(l, j);
  int n = nt * 16 + (l & 15);
  float v0 = head_fetch(n, k,     keyW, betaW, gateW, shiftW, gammaW, eraseW, addW);
  float v1 = head_fetch(n, k + 1, keyW, betaW, gateW, shiftW, gammaW, eraseW, addW);
  dst[(size_t)blk * 256 + l * 8 + j] = packbf(v0, v1);
}

// ---------------- LDS layout (dynamic) ----------------
#define OFF_CI   0        // ci bf16 A-tiles:   6 blocks * 1KB
#define OFF_HA   6144     // [h | rv] A-tiles: 10 blocks * 1KB (blk 8,9 = rv)
#define OFF_G    16384    // gates f32 [16][1024]
#define OFF_H    81920    // h f32 [16][256]
#define OFF_C    98304    // c f32 [16][256]
#define OFF_HD   114688   // head out f32 [16][224] (208 used)
#define OFF_K    129024   // keys f32 [16][64]
#define OFF_ER   133120   // erase [16][64]
#define OFF_AD   137216   // add   [16][64]
#define OFF_WP   141312   // w (persists across steps) [16][256]
#define OFF_W1   157696   // scratch [16][256]
#define OFF_W2   174080   // scratch [16][256]
#define OFF_RV   190464   // rv f32 [16][64]
#define OFF_SC   194560   // per-row scalars [16][8]: beta,gate,gamma,s0,s1,s2,kinv
#define OFF_RED  195072   // [16][17] reduce scratch
#define OFF_RED2 196160   // [16]
#define SMEM_BYTES 196224

// ---------------- persistent NTM kernel ----------------
__global__ void ntm_main_kernel(
    const int* __restrict__ x, const float* __restrict__ emb,
    const float* __restrict__ b_ih, const float* __restrict__ b_hh,
    const float* __restrict__ key_b, const float* __restrict__ beta_b,
    const float* __restrict__ gate_b, const float* __restrict__ shift_b,
    const float* __restrict__ gamma_b, const float* __restrict__ erase_b,
    const float* __restrict__ add_b, const float* __restrict__ out_b,
    const uint32_t* __restrict__ wihP, const uint32_t* __restrict__ whhP,
    const uint32_t* __restrict__ whdP, const uint32_t* __restrict__ woutP,
    float* __restrict__ memws, float* __restrict__ out) {
  extern __shared__ char smem[];
  uint32_t* ciP  = (uint32_t*)(smem + OFF_CI);
  uint32_t* haP  = (uint32_t*)(smem + OFF_HA);
  float* gts  = (float*)(smem + OFF_G);
  float* hS   = (float*)(smem + OFF_H);
  float* cS   = (float*)(smem + OFF_C);
  float* hd   = (float*)(smem + OFF_HD);
  float* keys = (float*)(smem + OFF_K);
  float* erS  = (float*)(smem + OFF_ER);
  float* adS  = (float*)(smem + OFF_AD);
  float* wp   = (float*)(smem + OFF_WP);
  float* w1   = (float*)(smem + OFF_W1);
  float* w2   = (float*)(smem + OFF_W2);
  float* rvf  = (float*)(smem + OFF_RV);
  float* scal = (float*)(smem + OFF_SC);
  float* red  = (float*)(smem + OFF_RED);
  float* red2 = (float*)(smem + OFF_RED2);

  const int g    = blockIdx.x;          // batch group: rows g*16 .. g*16+15
  const int tid  = threadIdx.x;
  const int wave = tid >> 5, lane = tid & 31;
  const int rr   = tid >> 4, sub = tid & 15;   // row / sub-lane partition

  // ---- init state: h=c=0, w=0, packed [h|rv]=0 ----
  for (int i = tid; i < RB * CC; i += THREADS) { hS[i] = 0.f; cS[i] = 0.f; }
  for (int i = tid; i < RB * NN; i += THREADS) wp[i] = 0.f;
  for (int i = tid; i < 10 * 256; i += THREADS) haP[i] = 0u;

  for (int t = 0; t < SS; ++t) {
    __syncthreads();
    // ---- A1: ci A-tiles 4,5 <- rv A-tiles (prev step) ----
    for (int i = 0; i < 2; ++i) {
      int d = tid + THREADS * i;                         // < 512
      ciP[1024 + d] = haP[2048 + d];
    }
    // ---- A2: pack embedded input into ci A-tiles 0..3 ----
    for (int i = 0; i < 4; ++i) {
      int d = tid + THREADS * i;                         // < 1024
      int kt = d >> 8, rem = d & 255, l = rem >> 3, j = rem & 7;
      int m = l & 15, k = kt * 32 + kmap(l, j);
      int row = x[(size_t)(g * RB + m) * SS + t];
      ciP[kt * 256 + rem] = packbf(emb[(size_t)row * DD + k],
                                   emb[(size_t)row * DD + k + 1]);
    }
    __syncthreads();

    // ---- B: gates[16][1024] = ci@WihT + h@WhhT (WMMA, 8 n-tiles/wave) ----
    for (int q = 0; q < 8; ++q) {
      int nt = wave * 8 + q;
      v8f acc = {0.f, 0.f, 0.f, 0.f, 0.f, 0.f, 0.f, 0.f};
      for (int kt = 0; kt < 6; ++kt)
        acc = wmma_step(acc, ciP + kt * 256, wihP + (size_t)(kt * 64 + nt) * 256, lane);
      for (int kt = 0; kt < 8; ++kt)
        acc = wmma_step(acc, haP + kt * 256, whhP + (size_t)(kt * 64 + nt) * 256, lane);
      int n = lane & 15, mb = (lane >> 4) << 3, cc = nt * 16 + n;
      #pragma unroll
      for (int v = 0; v < 8; ++v) gts[(mb + v) * 1024 + cc] = acc[v];
    }
    __syncthreads();

    // ---- C: LSTM pointwise (torch gate order i,f,g,o) ----
    for (int i = 0; i < 16; ++i) {
      int e = tid + THREADS * i;                         // < 4096
      int r = e >> 8, cc = e & 255;
      float gi = gts[r * 1024 + cc]       + b_ih[cc]       + b_hh[cc];
      float gf = gts[r * 1024 + 256 + cc] + b_ih[256 + cc] + b_hh[256 + cc];
      float gg = gts[r * 1024 + 512 + cc] + b_ih[512 + cc] + b_hh[512 + cc];
      float go = gts[r * 1024 + 768 + cc] + b_ih[768 + cc] + b_hh[768 + cc];
      float cs = sigf(gf) * cS[r * 256 + cc] + sigf(gi) * tanhf(gg);
      float hh = sigf(go) * tanhf(cs);
      cS[r * 256 + cc] = cs;
      hS[r * 256 + cc] = hh;
    }
    __syncthreads();
    // ---- C2: pack h into A-tiles 0..7 ----
    for (int i = 0; i < 8; ++i) {
      int d = tid + THREADS * i;                         // < 2048
      int kt = d >> 8, rem = d & 255, l = rem >> 3, j = rem & 7;
      int m = l & 15, k = kt * 32 + kmap(l, j);
      haP[kt * 256 + rem] = packbf(hS[m * 256 + k], hS[m * 256 + k + 1]);
    }
    __syncthreads();

    // ---- D: head projection [16][208] (13 n-tiles over 8 waves) ----
    {
      int cnt = (wave + 8 < 13) ? 2 : 1;                 // wave-uniform
      for (int ti = 0; ti < cnt; ++ti) {
        int nt = (ti == 0) ? wave : wave + 8;
        v8f acc = {0.f, 0.f, 0.f, 0.f, 0.f, 0.f, 0.f, 0.f};
        for (int kt = 0; kt < 8; ++kt)
          acc = wmma_step(acc, haP + kt * 256, whdP + (size_t)(kt * 13 + nt) * 256, lane);
        int n = lane & 15, mb = (lane >> 4) << 3;
        #pragma unroll
        for (int v = 0; v < 8; ++v) hd[(mb + v) * 224 + nt * 16 + n] = acc[v];
      }
    }
    __syncthreads();

    // ---- E: head nonlinearities + key norm ----
    {
      const float* hdr = hd + rr * 224;
      float ssq = 0.f;
      for (int q = 0; q < 4; ++q) {
        int m = sub * 4 + q;
        float kv = hdr[m] + key_b[m];
        keys[rr * MM + m] = kv;  ssq += kv * kv;
        erS[rr * MM + m] = sigf(hdr[70 + m] + erase_b[m]);
        adS[rr * MM + m] = tanhf(hdr[134 + m] + add_b[m]);
      }
      red[rr * 17 + sub] = ssq;
      __syncthreads();
      if (sub == 0) {
        float s = 0.f;
        for (int q = 0; q < 16; ++q) s += red[rr * 17 + q];
        scal[rr * 8 + 6] = 1.f / fmaxf(sqrtf(s), 1e-12f);        // key inv-norm
        scal[rr * 8 + 0] = softplusf(hdr[64] + beta_b[0]);       // beta
        scal[rr * 8 + 1] = sigf(hdr[65] + gate_b[0]);            // gate
        scal[rr * 8 + 2] = 1.f + softplusf(hdr[69] + gamma_b[0]);// gamma
        float e0 = hdr[66] + shift_b[0], e1 = hdr[67] + shift_b[1],
              e2 = hdr[68] + shift_b[2];
        float mx = fmaxf(e0, fmaxf(e1, e2));
        float x0 = __expf(e0 - mx), x1 = __expf(e1 - mx), x2 = __expf(e2 - mx);
        float si = 1.f / (x0 + x1 + x2);
        scal[rr * 8 + 3] = x0 * si; scal[rr * 8 + 4] = x1 * si; scal[rr * 8 + 5] = x2 * si;
      }
    }
    __syncthreads();

    // ---- F1: cosine-sim logits: w1[r][n] = beta * <mem_n/|.|, key/|.|> ----
    {
      int n = tid;                                       // 0..255 (one slot/thread)
      for (int r = 0; r < RB; ++r) {
        const float4* mr = (const float4*)(memws + ((size_t)(g * RB + r) * NN + n) * MM);
        const float4* kk = (const float4*)(keys + r * MM);
        float dot = 0.f, ssq = 0.f;
        #pragma unroll
        for (int m4 = 0; m4 < MM / 4; ++m4) {
          float4 mv = mr[m4], kv = kk[m4];
          dot += mv.x * kv.x + mv.y * kv.y + mv.z * kv.z + mv.w * kv.w;
          ssq += mv.x * mv.x + mv.y * mv.y + mv.z * mv.z + mv.w * mv.w;
        }
        float sim = dot * scal[r * 8 + 6] / fmaxf(sqrtf(ssq), 1e-12f);
        w1[r * 256 + n] = scal[r * 8 + 0] * sim;
      }
    }
    __syncthreads();

    // ---- F2/F3: softmax -> interpolate -> circular shift -> sharpen ----
    {
      float mx = -1e30f;
      for (int q = 0; q < 16; ++q)
        mx = fmaxf(mx, w1[rr * 256 + sub + 16 * q]);
      red[rr * 17 + sub] = mx;
      __syncthreads();
      if (sub == 0) {
        float m2 = red[rr * 17];
        for (int q = 1; q < 16; ++q) m2 = fmaxf(m2, red[rr * 17 + q]);
        red2[rr] = m2;
      }
      __syncthreads();
      float rowmax = red2[rr], s = 0.f;
      for (int q = 0; q < 16; ++q) {
        int n = sub + 16 * q;
        float e = __expf(w1[rr * 256 + n] - rowmax);
        w1[rr * 256 + n] = e;  s += e;
      }
      red[rr * 17 + sub] = s;
      __syncthreads();
      if (sub == 0) {
        float tt = 0.f;
        for (int q = 0; q < 16; ++q) tt += red[rr * 17 + q];
        red2[rr] = 1.f / tt;
      }
      __syncthreads();
      float inv = red2[rr], gt = scal[rr * 8 + 1];
      for (int q = 0; q < 16; ++q) {                     // w_g
        int n = sub + 16 * q;
        float wc = w1[rr * 256 + n] * inv;
        w2[rr * 256 + n] = gt * wc + (1.f - gt) * wp[rr * 256 + n];
      }
      __syncthreads();
      float s0 = scal[rr * 8 + 3], s1 = scal[rr * 8 + 4], s2v = scal[rr * 8 + 5];
      float gm = scal[rr * 8 + 2], ssum = 0.f;
      for (int q = 0; q < 16; ++q) {                     // conv + sharpen
        int n = sub + 16 * q;
        float wv = s0 * w2[rr * 256 + ((n + 1) & 255)] +
                   s1 * w2[rr * 256 + n] +
                   s2v * w2[rr * 256 + ((n + 255) & 255)];
        float wsh = (wv > 0.f) ? __expf(gm * __logf(wv)) : 0.f;
        w1[rr * 256 + n] = wsh;  ssum += wsh;
      }
      red[rr * 17 + sub] = ssum;
      __syncthreads();
      if (sub == 0) {
        float tt = 0.f;
        for (int q = 0; q < 16; ++q) tt += red[rr * 17 + q];
        red2[rr] = 1.f / tt;
      }
      __syncthreads();
      float invs = red2[rr];
      for (int q = 0; q < 16; ++q) {                     // final w (persists)
        int n = sub + 16 * q;
        wp[rr * 256 + n] = w1[rr * 256 + n] * invs;
      }
    }
    __syncthreads();

    // ---- F4/F5: fused read (rv = w@mem_old) + erase/add memory update ----
    for (int i = tid; i < RB * MM; i += THREADS) rvf[i] = 0.f;
    __syncthreads();
    {
      float acc[MM];
      #pragma unroll
      for (int m = 0; m < MM; ++m) acc[m] = 0.f;
      float4* base = (float4*)(memws + (size_t)(g * RB + rr) * NN * MM);
      const float4* er4 = (const float4*)(erS + rr * MM);
      const float4* ad4 = (const float4*)(adS + rr * MM);
      for (int q = 0; q < 16; ++q) {
        int n = sub + 16 * q;
        float wn = wp[rr * 256 + n];
        float4* mr = base + n * (MM / 4);
        #pragma unroll
        for (int m4 = 0; m4 < MM / 4; ++m4) {
          float4 mv = mr[m4];
          float4 e = er4[m4], a = ad4[m4];
          acc[m4 * 4 + 0] += wn * mv.x;  acc[m4 * 4 + 1] += wn * mv.y;
          acc[m4 * 4 + 2] += wn * mv.z;  acc[m4 * 4 + 3] += wn * mv.w;
          mv.x = mv.x * (1.f - wn * e.x) + wn * a.x;
          mv.y = mv.y * (1.f - wn * e.y) + wn * a.y;
          mv.z = mv.z * (1.f - wn * e.z) + wn * a.z;
          mv.w = mv.w * (1.f - wn * e.w) + wn * a.w;
          mr[m4] = mv;
        }
      }
      #pragma unroll
      for (int m = 0; m < MM; ++m) atomicAdd(&rvf[rr * MM + m], acc[m]);
    }
    __syncthreads();

    // ---- F6: pack rv into A-tiles 8,9 ----
    for (int i = 0; i < 2; ++i) {
      int d = tid + THREADS * i;                         // < 512
      int ktl = d >> 8, rem = d & 255, l = rem >> 3, j = rem & 7;
      int m = l & 15, k = ktl * 32 + kmap(l, j);
      haP[(8 + ktl) * 256 + rem] = packbf(rvf[m * MM + k], rvf[m * MM + k + 1]);
    }
    __syncthreads();

    // ---- G: logits[16][512] = [h|rv]@outWT + out_b (4 n-tiles/wave) ----
    for (int q = 0; q < 4; ++q) {
      int nt = wave * 4 + q;
      v8f acc = {0.f, 0.f, 0.f, 0.f, 0.f, 0.f, 0.f, 0.f};
      for (int kt = 0; kt < 10; ++kt)
        acc = wmma_step(acc, haP + kt * 256, woutP + (size_t)(kt * 32 + nt) * 256, lane);
      int n = lane & 15, mb = (lane >> 4) << 3, col = nt * 16 + n;
      float bias = out_b[col];
      #pragma unroll
      for (int v = 0; v < 8; ++v) {
        int b = g * RB + mb + v;
        out[((size_t)b * SS + t) * VV + col] = acc[v] + bias;
      }
    }
  }
}

// ---------------- workspace offsets (bytes) ----------------
#define WS_MEM   0                      // 64*256*64*4 = 4,194,304
#define WS_WIH   4194304                // 6*64 blocks * 1KB = 393,216
#define WS_WHH   4587520                // 8*64 * 1KB = 524,288
#define WS_WHD   5111808                // 8*13 * 1KB = 106,496
#define WS_WOUT  5218304                // 10*32 * 1KB = 327,680
// total required: 5,545,984 bytes

extern "C" void kernel_launch(void* const* d_in, const int* in_sizes, int n_in,
                              void* d_out, int out_size, void* d_ws, size_t ws_size,
                              hipStream_t stream) {
  (void)in_sizes; (void)n_in; (void)out_size; (void)ws_size;
  const int*   x       = (const int*)  d_in[0];
  const float* emb     = (const float*)d_in[1];
  const float* W_ih    = (const float*)d_in[2];
  const float* W_hh    = (const float*)d_in[3];
  const float* b_ih    = (const float*)d_in[4];
  const float* b_hh    = (const float*)d_in[5];
  const float* key_W   = (const float*)d_in[6];
  const float* key_b   = (const float*)d_in[7];
  const float* beta_W  = (const float*)d_in[8];
  const float* beta_b  = (const float*)d_in[9];
  const float* gate_W  = (const float*)d_in[10];
  const float* gate_b  = (const float*)d_in[11];
  const float* shift_W = (const float*)d_in[12];
  const float* shift_b = (const float*)d_in[13];
  const float* gamma_W = (const float*)d_in[14];
  const float* gamma_b = (const float*)d_in[15];
  const float* erase_W = (const float*)d_in[16];
  const float* erase_b = (const float*)d_in[17];
  const float* add_W   = (const float*)d_in[18];
  const float* add_b   = (const float*)d_in[19];
  const float* out_W   = (const float*)d_in[20];
  const float* out_b   = (const float*)d_in[21];
  const float* mem_init= (const float*)d_in[22];
  float* out = (float*)d_out;

  float*    memws = (float*)   ((char*)d_ws + WS_MEM);
  uint32_t* wihP  = (uint32_t*)((char*)d_ws + WS_WIH);
  uint32_t* whhP  = (uint32_t*)((char*)d_ws + WS_WHH);
  uint32_t* whdP  = (uint32_t*)((char*)d_ws + WS_WHD);
  uint32_t* woutP = (uint32_t*)((char*)d_ws + WS_WOUT);

  // Re-initialize mutable memory state + packed weights every call (deterministic).
  copy_mem_kernel<<<1024, THREADS, 0, stream>>>(mem_init, memws, BB * NN * MM);
  pack_w_kernel<<<6 * 64,  THREADS, 0, stream>>>(W_ih,  wihP, DD + MM, 4 * CC, 64);
  pack_w_kernel<<<8 * 64,  THREADS, 0, stream>>>(W_hh,  whhP, CC,      4 * CC, 64);
  pack_w_kernel<<<10 * 32, THREADS, 0, stream>>>(out_W, woutP, CC + MM, VV,    32);
  pack_head_kernel<<<8 * 13, THREADS, 0, stream>>>(key_W, beta_W, gate_W, shift_W,
                                                   gamma_W, erase_W, add_W, whdP);

  (void)hipFuncSetAttribute((const void*)ntm_main_kernel,
                            hipFuncAttributeMaxDynamicSharedMemorySize, SMEM_BYTES);
  ntm_main_kernel<<<NGRP, THREADS, SMEM_BYTES, stream>>>(
      x, emb, b_ih, b_hh, key_b, beta_b, gate_b, shift_b, gamma_b,
      erase_b, add_b, out_b, wihP, whhP, whdP, woutP, memws, out);
}
